// LMGRU_29540785062125
// MI455X (gfx1250) — matrix operand order repaired and checked by reading, not compiled, not run on hardware
//
#include <hip/hip_runtime.h>

// LMGRU on MI455X (gfx1250): bf16 WMMA (16x16x32) GEMMs with fp32 accumulate.
// Per step: S1 computes z/r/o gates + x-part of c gate; S2 computes the
// recurrent part of the c gate and the c/h pointwise update.
// Each wave owns a full M=64 x 16-column output strip (4 m-tile accumulators,
// single B stream -> weights read once per step from L2). The 64-row
// activation panel is staged into a double-buffered LDS pipeline via async
// global->LDS copies (ASYNCcnt) and consumed with conflict-free ds_load_b128.

typedef __attribute__((ext_vector_type(16))) __bf16 v16bf;
typedef __attribute__((ext_vector_type(8)))  __bf16 v8bf;
typedef __attribute__((ext_vector_type(8)))  float  v8f;

#define BATCH 64
#define SEQ   512
#define INS   512
#define HID   1024
#define COMB  1536
#define NCLS  1000

#define KC    256          // k-elements staged per chunk
#define AST   264          // padded LDS row stride (elems): 132 dwords = 4 mod 64
#define ABUF  (64 * AST)   // one LDS buffer (elems)

__device__ __forceinline__ v16bf bf_combine(v8bf lo, v8bf hi) {
  return __builtin_shufflevector(lo, hi, 0,1,2,3,4,5,6,7,8,9,10,11,12,13,14,15);
}

// B fragment (32x16 bf16): lane<16 holds column n=lane, K=0..15 consecutive;
// lanes>=16 hold K=16..31.  p points at (n, kbase+koffB) in row-major W[n][k].
__device__ __forceinline__ v16bf load_bfrag(const __bf16* p) {
  v8bf lo = *(const v8bf*)p;
  v8bf hi = *(const v8bf*)(p + 8);
  return bf_combine(lo, hi);
}

// A fragment (16x32 bf16) from LDS: p points at (row, kbase+koffA).
__device__ __forceinline__ v16bf load_afrag(const __bf16* p) {
  v8bf lo = *(const v8bf*)p;
  v8bf hi = *(const v8bf*)(p + 8);
  return bf_combine(lo, hi);
}

__device__ __forceinline__ float sigm(float x) {
  return 1.0f / (1.0f + __expf(-x));
}

// Issue async global->LDS copies for one 64-row x KC-element bf16 chunk.
// 2048 16-byte transfers, 8 per thread.  splitK is a multiple of KC, so a
// chunk never straddles the x/h boundary.
__device__ __forceinline__ void stage_issue(
    __bf16* dst, const __bf16* a0, int s0, const __bf16* a1, int s1,
    int splitK, int kbase, int tid) {
#pragma unroll
  for (int j = 0; j < 8; ++j) {
    const int e   = tid + 256 * j;
    const int m   = e >> 5;
    const int kin = (e & 31) * 8;
    const int kg  = kbase + kin;
    const __bf16* src = (kg < splitK)
        ? (a0 + (size_t)m * s0 + kg)
        : (a1 + (size_t)m * s1 + (kg - splitK));
    unsigned lds = (unsigned)(uintptr_t)(dst + m * AST + kin);
    unsigned long long ga = (unsigned long long)(uintptr_t)src;
    asm volatile("global_load_async_to_lds_b128 %0, %1, off"
                 :: "v"(lds), "v"(ga) : "memory");
  }
}

// Full M=64 strip GEMM: acc[mt] += A(panel) x B(w strip), double-buffered LDS.
__device__ __forceinline__ void gemm_strip(
    __bf16* As,                                   // LDS [2][ABUF]
    const __bf16* a0, int s0, const __bf16* a1, int s1, int splitK,
    int K, const __bf16* w, int tid, int lane, v8f acc[4]) {
  const int ln    = lane & 15;
  const int koffA = (lane & 16) ? 16 : 0;

  stage_issue(As, a0, s0, a1, s1, splitK, 0, tid);
  asm volatile("s_wait_asynccnt 0" ::: "memory");
  __syncthreads();

  for (int kbase = 0; kbase < K; kbase += KC) {
    const int cb = (kbase / KC) & 1;
    __bf16* cur = As + cb * ABUF;
    if (kbase + KC < K)      // overlap: fill the other buffer during compute
      stage_issue(As + (cb ^ 1) * ABUF, a0, s0, a1, s1, splitK, kbase + KC, tid);
    if (kbase + 2 * KC < K)
      __builtin_prefetch((const void*)(w + kbase + 2 * KC), 0, 1);

#pragma unroll
    for (int kl = 0; kl < KC; kl += 32) {
      v16bf bf = load_bfrag(w + kbase + kl);
      const __bf16* ap = cur + ln * AST + kl + koffA;
      v16bf A0 = load_afrag(ap);
      v16bf A1 = load_afrag(ap + 16 * AST);
      v16bf A2 = load_afrag(ap + 32 * AST);
      v16bf A3 = load_afrag(ap + 48 * AST);
      acc[0] = __builtin_amdgcn_wmma_f32_16x16x32_bf16(false, A0, false, bf, (short)0, acc[0], false, false);
      acc[1] = __builtin_amdgcn_wmma_f32_16x16x32_bf16(false, A1, false, bf, (short)0, acc[1], false, false);
      acc[2] = __builtin_amdgcn_wmma_f32_16x16x32_bf16(false, A2, false, bf, (short)0, acc[2], false, false);
      acc[3] = __builtin_amdgcn_wmma_f32_16x16x32_bf16(false, A3, false, bf, (short)0, acc[3], false, false);
    }
    // Our next-chunk copies had the whole compute phase to land; wait + sync
    // before anyone reads (or overwrites) the buffers.
    asm volatile("s_wait_asynccnt 0" ::: "memory");
    __syncthreads();
  }
}

// ---------------------------------------------------------------------------
// Prep kernels
// ---------------------------------------------------------------------------
__global__ __launch_bounds__(256) void conv_w_kernel(
    const float* __restrict__ wz, const float* __restrict__ wr,
    const float* __restrict__ wc, const float* __restrict__ wo,
    __bf16* __restrict__ Wbf) {
  size_t i = (size_t)blockIdx.x * blockDim.x + threadIdx.x;
  const size_t per = (size_t)HID * COMB;
  if (i >= 4 * per) return;
  int g = (int)(i / per);
  size_t j = i % per;
  const float* src = (g == 0) ? wz : (g == 1) ? wr : (g == 2) ? wc : wo;
  Wbf[i] = (__bf16)src[j];
}

__global__ __launch_bounds__(256) void conv_fc_kernel(
    const float* __restrict__ fcw, __bf16* __restrict__ fcbf) {
  size_t i = (size_t)blockIdx.x * blockDim.x + threadIdx.x;
  if (i >= (size_t)1024 * HID) return;
  int n = (int)(i / HID);
  int k = (int)(i % HID);
  fcbf[i] = (n < NCLS) ? (__bf16)fcw[(size_t)n * HID + k] : (__bf16)0.0f;
}

__global__ __launch_bounds__(256) void conv_x_kernel(
    const float* __restrict__ x, __bf16* __restrict__ xbf) {
  size_t i = (size_t)blockIdx.x * blockDim.x + threadIdx.x;
  if (i >= (size_t)BATCH * SEQ * INS) return;
  int k = (int)(i % INS);
  size_t r = i / INS;
  int t = (int)(r % SEQ);
  int b = (int)(r / SEQ);
  xbf[((size_t)t * BATCH + b) * INS + k] = (__bf16)x[i];
}

__global__ __launch_bounds__(256) void init_state_kernel(
    __bf16* __restrict__ hbf, float* __restrict__ cbuf) {
  int i = blockIdx.x * blockDim.x + threadIdx.x;
  if (i < BATCH * HID) { hbf[i] = (__bf16)0.0f; cbuf[i] = 0.0f; }
}

// ---------------------------------------------------------------------------
// S1: z, r, o gates (K=1536) + x-part of c gate (K=512).
// Grid: 32 blocks x 256 thr.  task = blockIdx.x>>3; wave -> one 16-col strip.
// ---------------------------------------------------------------------------
__global__ __launch_bounds__(256) void lmgru_gates_kernel(
    int t,
    const __bf16* __restrict__ xbf, const __bf16* __restrict__ hbf,
    const __bf16* __restrict__ Wbf,   // [4][HID][COMB] order z,r,c,o
    const float* __restrict__ bz, const float* __restrict__ br,
    const float* __restrict__ bc, const float* __restrict__ bo,
    const float* __restrict__ cbuf,
    float* __restrict__ zbuf, float* __restrict__ obuf,
    float* __restrict__ cpart, __bf16* __restrict__ rcbf) {
  __shared__ __bf16 As[2 * ABUF];

  const int tid  = threadIdx.x;
  const int lane = tid & 31;
  const int task = blockIdx.x >> 3;                  // 0=z 1=r 2=o 3=c_xpart
  const int nt   = (blockIdx.x & 7) * 8 + (tid >> 5);
  const int n0   = nt * 16;

  const int widx = (task == 0) ? 0 : (task == 1) ? 1 : (task == 2) ? 3 : 2;
  const __bf16* W = Wbf + (size_t)widx * HID * COMB;
  const float* bias = (task == 0) ? bz : (task == 1) ? br : (task == 2) ? bo : bc;
  const int K = (task == 3) ? INS : COMB;

  const int ln   = lane & 15;
  const int hi16 = (lane >> 4) & 1;
  const int koffB = hi16 ? 16 : 0;

  const __bf16* w  = W + (size_t)(n0 + ln) * COMB + koffB;
  const __bf16* a0 = xbf + (size_t)t * BATCH * INS;  // [64][INS] x panel, step t

  const float bv = bias[n0 + ln];
  v8f acc[4];
#pragma unroll
  for (int mt = 0; mt < 4; ++mt)
#pragma unroll
    for (int i = 0; i < 8; ++i) acc[mt][i] = bv;

  gemm_strip(As, a0, INS, hbf, HID, INS, K, w, tid, lane, acc);

#pragma unroll
  for (int mt = 0; mt < 4; ++mt) {
#pragma unroll
    for (int v = 0; v < 8; ++v) {
      const int m = mt * 16 + v + hi16 * 8;
      const size_t i = (size_t)m * HID + n0 + ln;
      const float a = acc[mt][v];
      if (task == 0)      zbuf[i]  = sigm(a);
      else if (task == 1) rcbf[i]  = (__bf16)(sigm(a) * cbuf[i]);
      else if (task == 2) obuf[i]  = sigm(a);
      else                cpart[i] = a;
    }
  }
}

// ---------------------------------------------------------------------------
// S2: c~ = tanh(cpart + (r*c) @ Wc_h^T); c' = z*c + (1-z)*c~; h' = o*tanh(c')
// Grid: 8 blocks x 256 thr; 64 waves, one 16-col strip each, K = 1024.
// ---------------------------------------------------------------------------
__global__ __launch_bounds__(256) void lmgru_update_kernel(
    const __bf16* __restrict__ rcbf, const __bf16* __restrict__ Wbf,
    const float* __restrict__ cpart, const float* __restrict__ zbuf,
    const float* __restrict__ obuf,
    float* __restrict__ cbuf, __bf16* __restrict__ hbf) {
  __shared__ __bf16 As[2 * ABUF];

  const int tid  = threadIdx.x;
  const int lane = tid & 31;
  const int nt   = blockIdx.x * 8 + (tid >> 5);
  const int n0   = nt * 16;

  const int ln   = lane & 15;
  const int hi16 = (lane >> 4) & 1;
  const int koffB = hi16 ? 16 : 0;

  const __bf16* Wc = Wbf + (size_t)2 * HID * COMB;   // c-gate weights
  const __bf16* w = Wc + (size_t)(n0 + ln) * COMB + INS + koffB;

  v8f acc[4];
#pragma unroll
  for (int mt = 0; mt < 4; ++mt)
#pragma unroll
    for (int v = 0; v < 8; ++v) {
      const int m = mt * 16 + v + hi16 * 8;
      acc[mt][v] = cpart[(size_t)m * HID + n0 + ln];
    }

  gemm_strip(As, rcbf, HID, rcbf, HID, HID, HID, w, tid, lane, acc);

#pragma unroll
  for (int mt = 0; mt < 4; ++mt) {
#pragma unroll
    for (int v = 0; v < 8; ++v) {
      const int m = mt * 16 + v + hi16 * 8;
      const size_t i = (size_t)m * HID + n0 + ln;
      const float z = zbuf[i], c = cbuf[i];
      const float cn = z * c + (1.0f - z) * tanhf(acc[mt][v]);
      cbuf[i] = cn;
      hbf[i] = (__bf16)(obuf[i] * tanhf(cn));
    }
  }
}

// ---------------------------------------------------------------------------
// Final FC: out[64][1000] = h @ fc_w^T + fc_b  (fc_w zero-padded to 1024 rows)
// Grid: 8 blocks x 256 thr; 64 waves, one 16-col strip each, K = 1024.
// ---------------------------------------------------------------------------
__global__ __launch_bounds__(256) void lmgru_fc_kernel(
    const __bf16* __restrict__ hbf, const __bf16* __restrict__ fcbf,
    const float* __restrict__ fcb, float* __restrict__ out) {
  __shared__ __bf16 As[2 * ABUF];

  const int tid  = threadIdx.x;
  const int lane = tid & 31;
  const int nt   = blockIdx.x * 8 + (tid >> 5);
  const int n0   = nt * 16;

  const int ln   = lane & 15;
  const int hi16 = (lane >> 4) & 1;
  const int koffB = hi16 ? 16 : 0;

  const __bf16* w = fcbf + (size_t)(n0 + ln) * HID + koffB;

  const int n = n0 + ln;
  const float bv = (n < NCLS) ? fcb[n] : 0.0f;
  v8f acc[4];
#pragma unroll
  for (int mt = 0; mt < 4; ++mt)
#pragma unroll
    for (int i = 0; i < 8; ++i) acc[mt][i] = bv;

  gemm_strip(As, hbf, HID, hbf, HID, HID, HID, w, tid, lane, acc);

  if (n < NCLS) {
#pragma unroll
    for (int mt = 0; mt < 4; ++mt)
#pragma unroll
      for (int v = 0; v < 8; ++v) {
        const int m = mt * 16 + v + hi16 * 8;
        out[(size_t)m * NCLS + n] = acc[mt][v];
      }
  }
}

// ---------------------------------------------------------------------------
// Host launcher
// ---------------------------------------------------------------------------
extern "C" void kernel_launch(void* const* d_in, const int* in_sizes, int n_in,
                              void* d_out, int out_size, void* d_ws, size_t ws_size,
                              hipStream_t stream) {
  (void)in_sizes; (void)n_in; (void)out_size; (void)ws_size;

  const float* x   = (const float*)d_in[0];
  const float* wz  = (const float*)d_in[1];
  const float* bz  = (const float*)d_in[2];
  const float* wr  = (const float*)d_in[3];
  const float* br  = (const float*)d_in[4];
  const float* wc  = (const float*)d_in[5];
  const float* bc  = (const float*)d_in[6];
  const float* wo  = (const float*)d_in[7];
  const float* bo  = (const float*)d_in[8];
  const float* fcw = (const float*)d_in[9];
  const float* fcb = (const float*)d_in[10];
  float* out = (float*)d_out;

  char* ws = (char*)d_ws;
  size_t off = 0;
  auto take = [&](size_t bytes) -> void* {
    void* p = (void*)(ws + off);
    off = (off + bytes + 255) & ~(size_t)255;
    return p;
  };

  __bf16* Wbf   = (__bf16*)take((size_t)4 * HID * COMB * 2);    // z,r,c,o bf16
  __bf16* fcbf  = (__bf16*)take((size_t)1024 * HID * 2);        // padded fc
  __bf16* xbf   = (__bf16*)take((size_t)SEQ * BATCH * INS * 2); // [T][B][K]
  __bf16* hbf   = (__bf16*)take((size_t)BATCH * HID * 2);
  __bf16* rcbf  = (__bf16*)take((size_t)BATCH * HID * 2);
  float*  cbuf  = (float*)take((size_t)BATCH * HID * 4);
  float*  zbuf  = (float*)take((size_t)BATCH * HID * 4);
  float*  obuf  = (float*)take((size_t)BATCH * HID * 4);
  float*  cpart = (float*)take((size_t)BATCH * HID * 4);

  // Prep: convert weights/x to bf16, zero state.
  {
    const size_t nW = (size_t)4 * HID * COMB;
    conv_w_kernel<<<(unsigned)((nW + 255) / 256), 256, 0, stream>>>(wz, wr, wc, wo, Wbf);
    const size_t nF = (size_t)1024 * HID;
    conv_fc_kernel<<<(unsigned)((nF + 255) / 256), 256, 0, stream>>>(fcw, fcbf);
    const size_t nX = (size_t)BATCH * SEQ * INS;
    conv_x_kernel<<<(unsigned)((nX + 255) / 256), 256, 0, stream>>>(x, xbf);
    init_state_kernel<<<(BATCH * HID + 255) / 256, 256, 0, stream>>>(hbf, cbuf);
  }

  // Recurrence: two kernels per step, stream order is the inter-step barrier.
  for (int t = 0; t < SEQ; ++t) {
    lmgru_gates_kernel<<<32, 256, 0, stream>>>(t, xbf, hbf, Wbf,
                                               bz, br, bc, bo,
                                               cbuf, zbuf, obuf, cpart, rcbf);
    lmgru_update_kernel<<<8, 256, 0, stream>>>(rcbf, Wbf, cpart, zbuf, obuf,
                                               cbuf, hbf);
  }

  // Classifier head.
  lmgru_fc_kernel<<<8, 256, 0, stream>>>(hbf, fcbf, fcb, out);
}
